// ISTIncrementalEncoder_10677288698327
// MI455X (gfx1250) — matrix operand order repaired and checked by reading, not compiled, not run on hardware
//
#include <hip/hip_runtime.h>
#include <hip/hip_bf16.h>
#include <math.h>

// ---------------------------------------------------------------------------
// IST incremental encoder forward for MI455X (gfx1250, wave32, WMMA).
// GEMMs: v_wmma_f32_16x16x32_f16 (f32 accumulate), one wave per 16x64 strip
// (4 N-tiles sharing one A fragment). K-step is phase-separated:
// [issue all loads] -> [convert] -> [4 x WMMA], so loads clause together and
// WMMA<->VALU WAR hazards are spaced out.
// ---------------------------------------------------------------------------

typedef __attribute__((ext_vector_type(16))) _Float16 v16h;
typedef __attribute__((ext_vector_type(8)))  float    v8f;

#define B_SZ 4096
#define F_SZ 512
#define H_SZ 256
#define T_SZ 64
#define G3H  768   // 3*H

__device__ __forceinline__ float sigmoidf_(float x) {
    return 1.0f / (1.0f + __expf(-x));
}

__device__ __forceinline__ void put4(v16h& d, int i, float4 f) {
    d[i + 0] = (_Float16)f.x;
    d[i + 1] = (_Float16)f.y;
    d[i + 2] = (_Float16)f.z;
    d[i + 3] = (_Float16)f.w;
}

// ---------------------------------------------------------------------------
// Strip GEMM:  C[M,N] = A[M,K] @ W[N,K]^T + bias[N]   (both operands fp32,
// converted to f16 in registers; W rows are contiguous in K).
// ---------------------------------------------------------------------------
__global__ __launch_bounds__(256)
void gemm_wmma_f16(const float* __restrict__ A, const float* __restrict__ W,
                   const float* __restrict__ bias, float* __restrict__ C,
                   int M, int N, int K)
{
    const int lane  = threadIdx.x & 31;
    const int wave  = threadIdx.x >> 5;
    const int strip = blockIdx.x * 8 + wave;       // 8 waves per block
    const int nsn   = N >> 6;                      // strips along N (64 wide)
    const int mt    = strip / nsn;
    const int nbase = (strip - mt * nsn) * 64;

    const int half = lane >> 4;                    // 0 or 1 (lane group)
    const int sub  = lane & 15;
    const int ka   = half * 16;                    // A: K sub-offset 0 / 16
    const int kb   = half * 16;                    // B: K sub-offset 0 / 16

    const float* Arow = A + (size_t)(mt * 16 + sub) * K + ka;
    const float* B0   = W + (size_t)(nbase + 0 * 16 + sub) * K + kb;
    const float* B1   = W + (size_t)(nbase + 1 * 16 + sub) * K + kb;
    const float* B2   = W + (size_t)(nbase + 2 * 16 + sub) * K + kb;
    const float* B3   = W + (size_t)(nbase + 3 * 16 + sub) * K + kb;

    v8f acc0 = {}, acc1 = {}, acc2 = {}, acc3 = {};

    for (int k0 = 0; k0 < K; k0 += 32) {
        // speculative prefetch of the streaming A operand, two K-steps ahead
        __builtin_prefetch(Arow + k0 + 64, 0, 0);

        // ---------------- phase 1: issue ALL loads (20 x b128) ------------
        const float4* ap  = reinterpret_cast<const float4*>(Arow + k0);
        const float4* ap2 = reinterpret_cast<const float4*>(Arow + k0 + 8);
        const float4* bp0 = reinterpret_cast<const float4*>(B0 + k0);
        const float4* bp1 = reinterpret_cast<const float4*>(B1 + k0);
        const float4* bp2 = reinterpret_cast<const float4*>(B2 + k0);
        const float4* bp3 = reinterpret_cast<const float4*>(B3 + k0);

        float4 ar0 = ap[0],  ar1 = ap[1];
        float4 ar2 = ap2[0], ar3 = ap2[1];
        float4 b00 = bp0[0], b01 = bp0[1], b02 = bp0[2], b03 = bp0[3];
        float4 b10 = bp1[0], b11 = bp1[1], b12 = bp1[2], b13 = bp1[3];
        float4 b20 = bp2[0], b21 = bp2[1], b22 = bp2[2], b23 = bp2[3];
        float4 b30 = bp3[0], b31 = bp3[1], b32 = bp3[2], b33 = bp3[3];

        // ---------------- phase 2: convert fp32 -> f16 fragments ----------
        // A 16x32: lane holds row `sub`; K = k0+ka..+7 in a[0..7],
        //          K = k0+ka+8..+15 in a[8..15]
        v16h a;
        put4(a, 0, ar0);  put4(a, 4, ar1);
        put4(a, 8, ar2);  put4(a, 12, ar3);
        // B 32x16: lane holds column `sub`; K = k0+kb..+15 in b[0..15]
        v16h f0, f1, f2, f3;
        put4(f0, 0, b00); put4(f0, 4, b01); put4(f0, 8, b02); put4(f0, 12, b03);
        put4(f1, 0, b10); put4(f1, 4, b11); put4(f1, 8, b12); put4(f1, 12, b13);
        put4(f2, 0, b20); put4(f2, 4, b21); put4(f2, 8, b22); put4(f2, 12, b23);
        put4(f3, 0, b30); put4(f3, 4, b31); put4(f3, 8, b32); put4(f3, 12, b33);

        // ---------------- phase 3: 4 back-to-back WMMAs -------------------
        acc0 = __builtin_amdgcn_wmma_f32_16x16x32_f16(false, a, false, f0,
                                                      (short)0, acc0, false, false);
        acc1 = __builtin_amdgcn_wmma_f32_16x16x32_f16(false, a, false, f1,
                                                      (short)0, acc1, false, false);
        acc2 = __builtin_amdgcn_wmma_f32_16x16x32_f16(false, a, false, f2,
                                                      (short)0, acc2, false, false);
        acc3 = __builtin_amdgcn_wmma_f32_16x16x32_f16(false, a, false, f3,
                                                      (short)0, acc3, false, false);
    }

    // ---- store: D vgpr r -> M = r + 8*half, N = sub (per sub-tile)
    const int m0 = mt * 16 + half * 8;
#pragma unroll
    for (int u = 0; u < 4; ++u) {
        const int n_out = nbase + u * 16 + sub;
        const float bvv = bias[n_out];
        const v8f av = (u == 0) ? acc0 : (u == 1) ? acc1 : (u == 2) ? acc2 : acc3;
#pragma unroll
        for (int r = 0; r < 8; ++r)
            C[(size_t)(m0 + r) * N + n_out] = av[r] + bvv;
    }
}

// ---------------------------------------------------------------------------
// One-off 256x256 transpose (Wc -> WcT) so the ctx GEMM can use the same
// contiguous [N,K] operand path as the gate GEMMs. LDS-tiled, conflict-free.
// ---------------------------------------------------------------------------
__global__ __launch_bounds__(256)
void transpose256(const float* __restrict__ src, float* __restrict__ dst)
{
    __shared__ float tile[16][17];
    const int tx = threadIdx.x & 15;
    const int ty = threadIdx.x >> 4;
    const int k0 = (blockIdx.x & 15) * 16;
    const int n0 = (blockIdx.x >> 4) * 16;
    tile[ty][tx] = src[(k0 + ty) * H_SZ + n0 + tx];
    __syncthreads();
    dst[(n0 + ty) * H_SZ + k0 + tx] = tile[tx][ty];
}

// ---------------------------------------------------------------------------
// GRU gate math. Needs gi and gh separately because n-gate = tanh(i_n + r*h_n).
// h_new written densely (attention uses it); h_out gets the masked update.
// ---------------------------------------------------------------------------
__global__ __launch_bounds__(256)
void gru_pointwise(const float* __restrict__ GI, const float* __restrict__ GH,
                   const float* __restrict__ h, const unsigned char* __restrict__ mask,
                   float* __restrict__ h_new, float* __restrict__ h_out)
{
    const int i = blockIdx.x * 256 + threadIdx.x;   // i in [0, B*H)
    const int b = i >> 8;                           // H == 256
    const int a = i & 255;
    const float* gi = GI + (size_t)b * G3H;
    const float* gh = GH + (size_t)b * G3H;
    const float r  = sigmoidf_(gi[a]            + gh[a]);
    const float z  = sigmoidf_(gi[a + H_SZ]     + gh[a + H_SZ]);
    const float n  = tanhf   (gi[a + 2 * H_SZ] + r * gh[a + 2 * H_SZ]);
    const float hv = h[i];
    const float hn = (1.0f - z) * n + z * hv;
    h_new[i] = hn;
    h_out[i] = mask[b] ? hn : hv;
}

// ---------------------------------------------------------------------------
// tok = tanh(tokens); hist = tok @ Wh + bh. Tiny (64x256): one block per row,
// token row broadcast from LDS, Wh column reads coalesced across threads.
// ---------------------------------------------------------------------------
__global__ __launch_bounds__(256)
void tok_hist(const float* __restrict__ tokens, const float* __restrict__ Wh,
              const float* __restrict__ bh,
              float* __restrict__ tok, float* __restrict__ hist)
{
    __shared__ float st[H_SZ];
    const int t = blockIdx.x;
    const int a = threadIdx.x;
    const float tv = tanhf(tokens[t * H_SZ + a]);
    st[a] = tv;
    tok[t * H_SZ + a] = tv;
    __syncthreads();
    float accv = bh[a];
#pragma unroll 4
    for (int k = 0; k < H_SZ; ++k)
        accv = fmaf(st[k], Wh[k * H_SZ + a], accv);
    hist[t * H_SZ + a] = accv;
}

// ---------------------------------------------------------------------------
// Per-batch-row attention + output epilogue. The hot loop: 64x256 tanh per
// block. 8 waves x 8 t-rows each; lane j-strided over A with ctx/v fragments
// in registers; wave32 xor-shuffle reduction for the dot with v.
// ---------------------------------------------------------------------------
__global__ __launch_bounds__(256)
void attention_out(const float* __restrict__ HIST, const float* __restrict__ CTX,
                   const float* __restrict__ TOK, const float* __restrict__ vvec,
                   const float* __restrict__ bv, const float* __restrict__ accumulator,
                   const unsigned char* __restrict__ mask, float* __restrict__ emb_out)
{
    __shared__ float s_w[T_SZ];
    const int b    = blockIdx.x;
    const int lane = threadIdx.x & 31;
    const int wave = threadIdx.x >> 5;

    float cp[8], vv[8];
#pragma unroll
    for (int j = 0; j < 8; ++j) {
        const int a = lane + 32 * j;
        cp[j] = CTX[(size_t)b * H_SZ + a];
        vv[j] = vvec[a];
    }

#pragma unroll
    for (int tt = 0; tt < 8; ++tt) {
        const int t = wave * 8 + tt;
        const float* hrow = HIST + t * H_SZ;
        float p = 0.0f;
#pragma unroll
        for (int j = 0; j < 8; ++j) {
            const float e = tanhf(hrow[lane + 32 * j] + cp[j]);
            p = fmaf(e, vv[j], p);
        }
#pragma unroll
        for (int off = 16; off > 0; off >>= 1)
            p += __shfl_xor(p, off, 32);
        if (lane == 0) s_w[t] = p + bv[0];
    }
    __syncthreads();

    if (threadIdx.x == 0) {                 // softmax over 64 scores, serial
        float mx = s_w[0];
        for (int t = 1; t < T_SZ; ++t) mx = fmaxf(mx, s_w[t]);
        float sum = 0.0f;
        for (int t = 0; t < T_SZ; ++t) { const float e = __expf(s_w[t] - mx); s_w[t] = e; sum += e; }
        const float inv = 1.0f / sum;
        for (int t = 0; t < T_SZ; ++t) s_w[t] *= inv;
    }
    __syncthreads();

    const int a = threadIdx.x;
    float e = 0.0f;
#pragma unroll 8
    for (int t = 0; t < T_SZ; ++t)
        e = fmaf(s_w[t], TOK[t * H_SZ + a], e);

    const float av = tanhf(accumulator[(size_t)b * H_SZ + a]);
    // DECAY = 1.0 - 0.9 = 0.1
    const float outv = mask[b] ? fmaf(av, 0.1f, e) : av;
    emb_out[(size_t)b * H_SZ + a] = outv;
}

// ---------------------------------------------------------------------------
extern "C" void kernel_launch(void* const* d_in, const int* in_sizes, int n_in,
                              void* d_out, int out_size, void* d_ws, size_t ws_size,
                              hipStream_t stream)
{
    const float* features = (const float*)d_in[0];   // [B,F]
    const float* h        = (const float*)d_in[1];   // [B,H]
    const float* accum    = (const float*)d_in[2];   // [B,H]
    // jnp bool array: 1 byte/element on device
    const unsigned char* mask = (const unsigned char*)d_in[3];  // [B]
    const float* tokens   = (const float*)d_in[4];   // [T,H]
    const float* W_ih     = (const float*)d_in[5];   // [3H,F]
    const float* W_hh     = (const float*)d_in[6];   // [3H,H]
    const float* b_ih     = (const float*)d_in[7];   // [3H]
    const float* b_hh     = (const float*)d_in[8];   // [3H]
    const float* Wh       = (const float*)d_in[9];   // [H,H]
    const float* bh       = (const float*)d_in[10];  // [H]
    const float* Wc       = (const float*)d_in[11];  // [H,H]
    const float* bc       = (const float*)d_in[12];  // [H]
    const float* v        = (const float*)d_in[13];  // [H]
    const float* bv       = (const float*)d_in[14];  // [1]

    float* out_emb = (float*)d_out;                        // [B,H] embedding_out
    float* out_h   = out_emb + (size_t)B_SZ * H_SZ;        // [B,H] h_out

    // workspace (fp32): GI | GH | h_new | ctx | tok | hist | WcT  (~34 MB)
    float* ws   = (float*)d_ws;
    float* GI   = ws;
    float* GH   = GI   + (size_t)B_SZ * G3H;
    float* HN   = GH   + (size_t)B_SZ * G3H;
    float* CTX  = HN   + (size_t)B_SZ * H_SZ;
    float* TOK  = CTX  + (size_t)B_SZ * H_SZ;
    float* HIST = TOK  + (size_t)T_SZ * H_SZ;
    float* WCT  = HIST + (size_t)T_SZ * H_SZ;

    // 0) WcT = Wc^T (one-off, so ctx GEMM uses the contiguous operand path)
    transpose256<<<256, 256, 0, stream>>>(Wc, WCT);
    // 1) GI = features @ W_ih^T + b_ih    [4096 x 768], K=512
    gemm_wmma_f16<<<(B_SZ / 16) * (G3H / 64) / 8, 256, 0, stream>>>(
        features, W_ih, b_ih, GI, B_SZ, G3H, F_SZ);
    // 2) GH = h @ W_hh^T + b_hh           [4096 x 768], K=256
    gemm_wmma_f16<<<(B_SZ / 16) * (G3H / 64) / 8, 256, 0, stream>>>(
        h, W_hh, b_hh, GH, B_SZ, G3H, H_SZ);
    // 3) GRU gates -> h_new (dense) + h_out (masked)
    gru_pointwise<<<(B_SZ * H_SZ) / 256, 256, 0, stream>>>(GI, GH, h, mask, HN, out_h);
    // 4) tok = tanh(tokens); hist = tok @ Wh + bh
    tok_hist<<<T_SZ, 256, 0, stream>>>(tokens, Wh, bh, TOK, HIST);
    // 5) ctx = h_new @ WcT^T + bc         [4096 x 256], K=256
    gemm_wmma_f16<<<(B_SZ / 16) * (H_SZ / 64) / 8, 256, 0, stream>>>(
        HN, WCT, bc, CTX, B_SZ, H_SZ, H_SZ);
    // 6) attention scores + softmax + weights@tok + decay epilogue
    attention_out<<<B_SZ, 256, 0, stream>>>(HIST, CTX, TOK, v, bv, accum, mask, out_emb);
}
